// SBC_Net_19911468384452
// MI455X (gfx1250) — matrix operand order repaired and checked by reading, not compiled, not run on hardware
//
#include <hip/hip_runtime.h>
#include <hip/hip_bf16.h>

typedef __bf16 bhalf;
typedef __attribute__((ext_vector_type(16))) __bf16 v16bf;
typedef __attribute__((ext_vector_type(8)))  __bf16 v8bf;
typedef __attribute__((ext_vector_type(8)))  float  v8f;
typedef __attribute__((ext_vector_type(4))) unsigned int u32x4;
typedef __attribute__((ext_vector_type(8))) unsigned int u32x8;

#define DIMC   256
#define NTOK   2048
#define BATCH  2
#define MTOK   4096   // BATCH*NTOK
#define HEADS  8
#define PPTS   16
#define HDIM   32

__device__ __forceinline__ float gelu_f(float x) {
    return 0.5f * x * (1.0f + erff(x * 0.70710678118654752f));
}

// ---------------------------------------------------------------------------
// Pack x (B,DIM,N) -> xt bf16 [M,256], concat(xt,xyz) bf16 [M,288] (zero pad),
// and xt f32 [M,256] for the residual.
// ---------------------------------------------------------------------------
__global__ void pack_x_kernel(const float* __restrict__ x, const float* __restrict__ xyz,
                              bhalf* __restrict__ A1, bhalf* __restrict__ Axt,
                              float* __restrict__ xtf) {
    int idx = blockIdx.x * blockDim.x + threadIdx.x;
    if (idx >= MTOK * 288) return;
    int t = idx / 288, c = idx % 288;
    int b = t / NTOK, n = t % NTOK;
    float v = 0.f;
    if (c < 256)      v = x[((size_t)b * 256 + c) * NTOK + n];
    else if (c < 259) v = xyz[(size_t)t * 3 + (c - 256)];
    A1[idx] = (bhalf)v;
    if (c < 256) {
        Axt[(size_t)t * 256 + c] = (bhalf)v;
        xtf[(size_t)t * 256 + c] = v;
    }
}

// ---------------------------------------------------------------------------
// Pack weight fp32 [Kin, Nout] -> bf16 transposed [Nout, Kp] (zero pad K).
// ---------------------------------------------------------------------------
__global__ void pack_w_kernel(const float* __restrict__ W, bhalf* __restrict__ Wt,
                              int Kin, int Kp, int Nout) {
    int idx = blockIdx.x * blockDim.x + threadIdx.x;
    if (idx >= Nout * Kp) return;
    int nn = idx / Kp, k = idx % Kp;
    Wt[idx] = (k < Kin) ? (bhalf)W[(size_t)k * Nout + nn] : (bhalf)0.0f;
}

// ---------------------------------------------------------------------------
// Generic bf16 WMMA GEMM:  C[M,Nout] = A[M,Kp] @ Wt^T  (+bias, +act)
//   A row-major bf16 [M,Kp], Wt bf16 [Nout,Kp] (i.e. W transposed).
//   One 16x64 output tile per wave (A-fragment reused by 4 WMMAs per k-step),
//   8 waves per block.  act: 0 = none, 1 = exact GELU.  Cf/Cb nullable.
// ---------------------------------------------------------------------------
__global__ void gemm_wmma_kernel(const bhalf* __restrict__ A, const bhalf* __restrict__ Wt,
                                 const float* __restrict__ bias,
                                 float* __restrict__ Cf, bhalf* __restrict__ Cb,
                                 int Mrows, int Nout, int Kp, int act) {
    const int wave = threadIdx.x >> 5;
    const int lane = threadIdx.x & 31;
    const int ntn  = Nout >> 6;                 // 64-wide N tiles
    const int tile = blockIdx.x * 8 + wave;
    if (tile >= (Mrows >> 4) * ntn) return;
    const int tm = tile / ntn, tn = tile % ntn;
    const int hf  = lane >> 4;                  // half-wave
    const int l15 = lane & 15;
    const bhalf* Ap = A  + (size_t)(tm * 16 + l15) * Kp + hf * 16;
    const bhalf* Bp = Wt + (size_t)(tn * 64 + l15) * Kp + hf * 16;

    // warm L2 for this tile's row streams (global_prefetch_b8)
    __builtin_prefetch(Ap, 0, 3);
#pragma unroll
    for (int s = 0; s < 4; ++s) __builtin_prefetch(Bp + (size_t)(s * 16) * Kp, 0, 3);

    v8f acc[4] = {v8f{}, v8f{}, v8f{}, v8f{}};
    for (int kb = 0; kb < Kp; kb += 32) {
        v8bf alo = *(const v8bf*)(Ap + kb);        // K = hf*16 + 0..7
        v8bf ahi = *(const v8bf*)(Ap + kb + 8);    // K = hf*16 + 8..15
        v16bf av;
#pragma unroll
        for (int i = 0; i < 8; ++i) { av[i] = alo[i]; av[i + 8] = ahi[i]; }
#pragma unroll
        for (int s = 0; s < 4; ++s) {
            v16bf bv = *(const v16bf*)(Bp + (size_t)(s * 16) * Kp + kb);  // K = hf*16 + 0..15
            acc[s] = __builtin_amdgcn_wmma_f32_16x16x32_bf16(false, av, false, bv,
                                                             (short)0, acc[s], false, false);
        }
    }
#pragma unroll
    for (int s = 0; s < 4; ++s) {
        const int col = tn * 64 + s * 16 + l15;
        const float bb = bias ? bias[col] : 0.f;
#pragma unroll
        for (int r = 0; r < 8; ++r) {
            int row = tm * 16 + hf * 8 + r;
            float v = acc[s][r] + bb;
            if (act == 1) v = gelu_f(v);
            if (Cf) Cf[(size_t)row * Nout + col] = v;
            if (Cb) Cb[(size_t)row * Nout + col] = (bhalf)v;
        }
    }
}

// ---------------------------------------------------------------------------
// LayerNorm over C=256 (one row per 256-thread block), optional residual,
// optional GELU, f32 and/or bf16 outputs, optional transposed final store.
// ---------------------------------------------------------------------------
__global__ void ln_kernel(const float* __restrict__ X, const float* __restrict__ Res,
                          const float* __restrict__ g, const float* __restrict__ bb,
                          float* __restrict__ Of, bhalf* __restrict__ Ob,
                          int act, int transpose_out) {
    const int t = blockIdx.x, c = threadIdx.x;
    __shared__ float red[256];
    float v = X[(size_t)t * DIMC + c];
    if (Res) v += Res[(size_t)t * DIMC + c];
    red[c] = v; __syncthreads();
    for (int s = 128; s > 0; s >>= 1) { if (c < s) red[c] += red[c + s]; __syncthreads(); }
    float mean = red[0] * (1.f / DIMC); __syncthreads();
    float d = v - mean;
    red[c] = d * d; __syncthreads();
    for (int s = 128; s > 0; s >>= 1) { if (c < s) red[c] += red[c + s]; __syncthreads(); }
    float var = red[0] * (1.f / DIMC);
    float y = d * rsqrtf(var + 1e-5f) * g[c] + bb[c];
    if (act == 1) y = gelu_f(y);
    if (Of) {
        size_t oi = transpose_out
                  ? ((size_t)(t / NTOK) * DIMC + c) * NTOK + (t % NTOK)
                  : (size_t)t * DIMC + c;
        Of[oi] = y;
    }
    if (Ob) Ob[(size_t)t * DIMC + c] = (bhalf)y;
}

// ---------------------------------------------------------------------------
// off = tanh(off_pre)*|scale|;  sampled = xyz + off   (layout [M, H*P*3])
// ---------------------------------------------------------------------------
__global__ void offset_kernel(const float* __restrict__ off_pre, const float* __restrict__ xyz,
                              const float* __restrict__ scale,
                              float* __restrict__ off, float* __restrict__ sampled) {
    int idx = blockIdx.x * blockDim.x + threadIdx.x;
    if (idx >= MTOK * 384) return;
    int t = idx / 384, j = idx % 384;
    int k = j % 3;
    float v = tanhf(off_pre[idx]) * fabsf(scale[0]);
    off[idx]     = v;
    sampled[idx] = xyz[(size_t)t * 3 + k] + v;
}

// ---------------------------------------------------------------------------
// Fused: top-3 NN search + inverse-distance interp + Q.K attention logits +
// rel-pos-bias MLP + softmax(P=16) + V reduction  ->  ctx bf16 [M, 256].
// Block = 256 threads (8 waves), 2 tokens per block.  The per-batch point
// cloud (2048 x 3 f32 = 24 KB) is DMA'd into LDS with one TENSOR_LOAD_TO_LDS
// issued by wave 0 (D# groups built in SGPR tuples), completed with
// s_wait_tensorcnt, published by the workgroup barrier.
// ---------------------------------------------------------------------------
__global__ void attn_kernel(const float* __restrict__ sampled, const float* __restrict__ off,
                            const float* __restrict__ xyz,
                            const float* __restrict__ Qf, const float* __restrict__ Kf,
                            const float* __restrict__ Vf,
                            const float* __restrict__ rp_w1, const float* __restrict__ rp_b1,
                            const float* __restrict__ rp_g,  const float* __restrict__ rp_bb,
                            const float* __restrict__ rp_w2, const float* __restrict__ rp_b2,
                            bhalf* __restrict__ ctx_bf) {
    __shared__ float s_xyz[NTOK * 3];       // 24 KB: candidate point cloud
    __shared__ float s_contrib[256 * 32];   // 32 KB: per-lane weighted V vectors
    __shared__ float s_rp[448];             // rel-pos MLP params

    const int tid  = threadIdx.x;
    const int b    = blockIdx.x / (NTOK / 2);
    const int n0   = (blockIdx.x % (NTOK / 2)) * 2;

    // ---- TDM: one DMA of the whole point cloud into LDS (wave 0 issues) ----
    if (tid < 32) {
        unsigned long long ga = (unsigned long long)(const void*)(xyz + (size_t)b * NTOK * 3);
        unsigned int lds = (unsigned int)(size_t)&s_xyz[0];   // AS3 offset = addr[31:0]
        u32x4 g0;
        g0[0] = 1u;                                   // count=1 (valid), no gather
        g0[1] = lds;                                  // lds_addr (bytes)
        g0[2] = (unsigned int)ga;                     // global_addr[31:0]
        g0[3] = (unsigned int)(ga >> 32) | (2u << 30); // global_addr[56:32] | type=2
        u32x8 g1;
        g1[0] = 2u << 16;                             // data_size = 4 B, mask = 0
        g1[1] = (6144u & 0xFFFFu) << 16;              // tensor_dim0[15:0]
        g1[2] = (6144u >> 16) | (1u << 16);           // tensor_dim0[31:16] | tensor_dim1[15:0]
        g1[3] = (6144u << 16);                        // tensor_dim1[31:16] | tile_dim0 = 6144
        g1[4] = 1u;                                   // tile_dim1 = 1, tile_dim2 = 0
        g1[5] = 6144u;                                // tensor_dim0_stride[31:0]
        g1[6] = 0u;
        g1[7] = 0u;
        asm volatile("tensor_load_to_lds %0, %1" :: "s"(g0), "s"(g1) : "memory");
        __builtin_amdgcn_s_wait_tensorcnt(0);
    }
    if (tid < 192) s_rp[tid] = rp_w1[tid];
    if (tid < 64) {
        s_rp[192 + tid] = rp_b1[tid];
        s_rp[256 + tid] = rp_g[tid];
        s_rp[320 + tid] = rp_bb[tid];
        s_rp[384 + tid] = rp_w2[tid];
    }
    __syncthreads();

    const int wave    = tid >> 5;
    const int lane    = tid & 31;
    const int n_local = wave >> 2;
    const int h       = (wave & 3) * 2 + (lane >> 4);
    const int p       = lane & 15;
    const int n       = n0 + n_local;
    const int t       = b * NTOK + n;

    // query point
    const size_t qoff = (size_t)t * 384 + (size_t)(h * PPTS + p) * 3;
    const float qx = sampled[qoff + 0];
    const float qy = sampled[qoff + 1];
    const float qz = sampled[qoff + 2];

    // ---- top-3 nearest neighbors over 2048 candidates (LDS broadcast) ----
    float e0 = 3.4e38f, e1 = 3.4e38f, e2 = 3.4e38f;
    int   j0 = 0, j1 = 0, j2 = 0;
    for (int i = 0; i < NTOK; ++i) {
        float dx = qx - s_xyz[i * 3 + 0];
        float dy = qy - s_xyz[i * 3 + 1];
        float dz = qz - s_xyz[i * 3 + 2];
        float d  = dx * dx + dy * dy + dz * dz;
        if (d < e0)      { e2 = e1; j2 = j1; e1 = e0; j1 = j0; e0 = d; j0 = i; }
        else if (d < e1) { e2 = e1; j2 = j1; e1 = d;  j1 = i; }
        else if (d < e2) { e2 = d;  j2 = i; }
    }
    float w0 = 1.f / (e0 + 1e-8f), w1 = 1.f / (e1 + 1e-8f), w2 = 1.f / (e2 + 1e-8f);
    float ws = w0 + w1 + w2;
    w0 /= ws; w1 /= ws; w2 /= ws;

    // ---- attention logit = (Q . interp_K) * HD^-0.5 ----
    const float4* qrow = (const float4*)(Qf + (size_t)t * DIMC + h * HDIM);
    float4 q[8];
#pragma unroll
    for (int r = 0; r < 8; ++r) q[r] = qrow[r];

    const int   rowi[3] = { b * NTOK + j0, b * NTOK + j1, b * NTOK + j2 };
    const float wgt[3]  = { w0, w1, w2 };
    float dot = 0.f;
#pragma unroll
    for (int j = 0; j < 3; ++j) {
        const float4* kr = (const float4*)(Kf + (size_t)rowi[j] * DIMC + h * HDIM);
        float dj = 0.f;
#pragma unroll
        for (int r = 0; r < 8; ++r) {
            float4 kv = kr[r];
            dj += q[r].x * kv.x + q[r].y * kv.y + q[r].z * kv.z + q[r].w * kv.w;
        }
        dot += wgt[j] * dj;
    }
    float logit = dot * 0.17677669529663687f;   // 32^-0.5

    // ---- rel-pos bias MLP:  rel = -off ;  (3->64) LN gelu -> (64->1) ----
    const float rx = -off[qoff + 0], ry = -off[qoff + 1], rz = -off[qoff + 2];
    float s1 = 0.f, s2 = 0.f;
    for (int j = 0; j < 64; ++j) {
        float tj = rx * s_rp[j] + ry * s_rp[64 + j] + rz * s_rp[128 + j] + s_rp[192 + j];
        s1 += tj; s2 += tj * tj;
    }
    float mean = s1 * (1.f / 64.f);
    float var  = s2 * (1.f / 64.f) - mean * mean;
    float inv  = rsqrtf(var + 1e-5f);
    float rb   = rp_b2[0];
    for (int j = 0; j < 64; ++j) {
        float tj = rx * s_rp[j] + ry * s_rp[64 + j] + rz * s_rp[128 + j] + s_rp[192 + j];
        float u  = (tj - mean) * inv * s_rp[256 + j] + s_rp[320 + j];
        rb += gelu_f(u) * s_rp[384 + j];
    }

    // ---- softmax over P=16 (half-wave shuffle, wave32) ----
    float e = logit + rb;
    float m = e;
    for (int o = 8; o >= 1; o >>= 1) m = fmaxf(m, __shfl_xor(m, o, 16));
    float ex = expf(e - m);
    float sm = ex;
    for (int o = 8; o >= 1; o >>= 1) sm += __shfl_xor(sm, o, 16);
    float a = ex / sm;

    // ---- weighted V contribution ----
    float4 acc4[8];
#pragma unroll
    for (int r = 0; r < 8; ++r) acc4[r] = make_float4(0.f, 0.f, 0.f, 0.f);
#pragma unroll
    for (int j = 0; j < 3; ++j) {
        float cj = a * wgt[j];
        const float4* vr = (const float4*)(Vf + (size_t)rowi[j] * DIMC + h * HDIM);
#pragma unroll
        for (int r = 0; r < 8; ++r) {
            float4 vv = vr[r];
            acc4[r].x += cj * vv.x; acc4[r].y += cj * vv.y;
            acc4[r].z += cj * vv.z; acc4[r].w += cj * vv.w;
        }
    }
    float4* sc = (float4*)&s_contrib[tid * 32];
#pragma unroll
    for (int r = 0; r < 8; ++r) sc[r] = acc4[r];
    __syncthreads();

    // ---- reduce over P and emit ctx bf16 ----
    for (int item = tid; item < 512; item += 256) {
        int nl = item >> 8, rem = item & 255;
        int hh = rem >> 5, cc = rem & 31;
        int wsrc = nl * 4 + (hh >> 1);
        int lbase = (hh & 1) << 4;
        float s = 0.f;
        for (int pp = 0; pp < 16; ++pp)
            s += s_contrib[(wsrc * 32 + lbase + pp) * 32 + cc];
        int tt = b * NTOK + n0 + nl;
        ctx_bf[(size_t)tt * DIMC + hh * HDIM + cc] = (bhalf)s;
    }
}

// ---------------------------------------------------------------------------
extern "C" void kernel_launch(void* const* d_in, const int* in_sizes, int n_in,
                              void* d_out, int out_size, void* d_ws, size_t ws_size,
                              hipStream_t stream) {
    (void)in_sizes; (void)n_in; (void)out_size; (void)ws_size;
    const float* x        = (const float*)d_in[0];
    const float* xyz      = (const float*)d_in[1];
    const float* scale    = (const float*)d_in[2];
    const float* off_w1   = (const float*)d_in[3];
    const float* off_b1   = (const float*)d_in[4];
    const float* off_ln_g = (const float*)d_in[5];
    const float* off_ln_b = (const float*)d_in[6];
    const float* off_w2   = (const float*)d_in[7];
    const float* off_b2   = (const float*)d_in[8];
    const float* off_w3   = (const float*)d_in[9];
    const float* off_b3   = (const float*)d_in[10];
    const float* wq       = (const float*)d_in[11];
    const float* wk       = (const float*)d_in[12];
    const float* wv       = (const float*)d_in[13];
    const float* wo       = (const float*)d_in[14];
    const float* rp_w1    = (const float*)d_in[15];
    const float* rp_b1    = (const float*)d_in[16];
    const float* rp_ln_g  = (const float*)d_in[17];
    const float* rp_ln_b  = (const float*)d_in[18];
    const float* rp_w2    = (const float*)d_in[19];
    const float* rp_b2    = (const float*)d_in[20];
    const float* ffn_w1   = (const float*)d_in[21];
    const float* ffn_b1   = (const float*)d_in[22];
    const float* ffn_w2   = (const float*)d_in[23];
    const float* ffn_b2   = (const float*)d_in[24];
    const float* n1_g     = (const float*)d_in[25];
    const float* n1_b     = (const float*)d_in[26];
    const float* n2_g     = (const float*)d_in[27];
    const float* n2_b     = (const float*)d_in[28];
    float* out            = (float*)d_out;

    // ---- workspace carve-up (256-B aligned) ----
    size_t cur = 0;
    char* base = (char*)d_ws;
    auto alloc = [&](size_t bytes) -> char* {
        char* r = base + cur;
        cur += (bytes + 255) & ~(size_t)255;
        return r;
    };
    bhalf* A1_bf    = (bhalf*)alloc((size_t)MTOK * 288 * 2);
    bhalf* Axt_bf   = (bhalf*)alloc((size_t)MTOK * 256 * 2);
    float* xt_f32   = (float*)alloc((size_t)MTOK * 256 * 4);
    bhalf* Wt1      = (bhalf*)alloc((size_t)256 * 288 * 2);
    bhalf* Wt2      = (bhalf*)alloc((size_t)128 * 256 * 2);
    bhalf* Wt3      = (bhalf*)alloc((size_t)384 * 128 * 2);
    bhalf* Wtq      = (bhalf*)alloc((size_t)256 * 256 * 2);
    bhalf* Wtk      = (bhalf*)alloc((size_t)256 * 256 * 2);
    bhalf* Wtv      = (bhalf*)alloc((size_t)256 * 256 * 2);
    bhalf* Wto      = (bhalf*)alloc((size_t)256 * 256 * 2);
    bhalf* Wtf1     = (bhalf*)alloc((size_t)1024 * 256 * 2);
    bhalf* Wtf2     = (bhalf*)alloc((size_t)256 * 1024 * 2);
    float* h_pre    = (float*)alloc((size_t)MTOK * 256 * 4);
    bhalf* h_bf     = (bhalf*)alloc((size_t)MTOK * 256 * 2);
    bhalf* h2_bf    = (bhalf*)alloc((size_t)MTOK * 128 * 2);
    float* off_pre  = (float*)alloc((size_t)MTOK * 384 * 4);
    float* off_v    = (float*)alloc((size_t)MTOK * 384 * 4);
    float* sampled  = (float*)alloc((size_t)MTOK * 384 * 4);
    float* Qf       = (float*)alloc((size_t)MTOK * 256 * 4);
    float* Kf       = (float*)alloc((size_t)MTOK * 256 * 4);
    float* Vf       = (float*)alloc((size_t)MTOK * 256 * 4);
    bhalf* ctx_bf   = (bhalf*)alloc((size_t)MTOK * 256 * 2);
    float* attn_out = (float*)alloc((size_t)MTOK * 256 * 4);
    float* y1       = (float*)alloc((size_t)MTOK * 256 * 4);
    bhalf* y1_bf    = (bhalf*)alloc((size_t)MTOK * 256 * 2);
    bhalf* f1_bf    = (bhalf*)alloc((size_t)MTOK * 1024 * 2);
    float* f2       = (float*)alloc((size_t)MTOK * 256 * 4);

    auto packw = [&](const float* W, bhalf* Wt, int Kin, int Kp, int Nout) {
        int nthr = Nout * Kp;
        pack_w_kernel<<<(nthr + 255) / 256, 256, 0, stream>>>(W, Wt, Kin, Kp, Nout);
    };
    auto gemm = [&](const bhalf* A, const bhalf* Wt, const float* bias,
                    float* Cf, bhalf* Cb, int Mr, int No, int Kp, int act) {
        int tiles = (Mr / 16) * (No / 64);
        gemm_wmma_kernel<<<(tiles + 7) / 8, 256, 0, stream>>>(A, Wt, bias, Cf, Cb, Mr, No, Kp, act);
    };

    // ---- pack activations + weights (fp32 -> bf16, transpose, pad) ----
    pack_x_kernel<<<(MTOK * 288 + 255) / 256, 256, 0, stream>>>(x, xyz, A1_bf, Axt_bf, xt_f32);
    packw(off_w1, Wt1, 259, 288, 256);
    packw(off_w2, Wt2, 256, 256, 128);
    packw(off_w3, Wt3, 128, 128, 384);
    packw(wq, Wtq, 256, 256, 256);
    packw(wk, Wtk, 256, 256, 256);
    packw(wv, Wtv, 256, 256, 256);
    packw(wo, Wto, 256, 256, 256);
    packw(ffn_w1, Wtf1, 256, 256, 1024);
    packw(ffn_w2, Wtf2, 1024, 1024, 256);

    // ---- offset MLP ----
    gemm(A1_bf, Wt1, off_b1, h_pre, nullptr, MTOK, 256, 288, 0);
    ln_kernel<<<MTOK, 256, 0, stream>>>(h_pre, nullptr, off_ln_g, off_ln_b,
                                        nullptr, h_bf, /*act=*/1, 0);
    gemm(h_bf, Wt2, off_b2, nullptr, h2_bf, MTOK, 128, 256, 1);
    gemm(h2_bf, Wt3, off_b3, off_pre, nullptr, MTOK, 384, 128, 0);
    offset_kernel<<<(MTOK * 384 + 255) / 256, 256, 0, stream>>>(off_pre, xyz, scale,
                                                                off_v, sampled);

    // ---- Q/K/V projections ----
    gemm(Axt_bf, Wtq, nullptr, Qf, nullptr, MTOK, 256, 256, 0);
    gemm(Axt_bf, Wtk, nullptr, Kf, nullptr, MTOK, 256, 256, 0);
    gemm(Axt_bf, Wtv, nullptr, Vf, nullptr, MTOK, 256, 256, 0);

    // ---- fused NN-search + interp + attention + softmax + V reduce ----
    attn_kernel<<<BATCH * (NTOK / 2), 256, 0, stream>>>(sampled, off_v, xyz, Qf, Kf, Vf,
                                                        rp_w1, rp_b1, rp_ln_g, rp_ln_b,
                                                        rp_w2, rp_b2, ctx_bf);

    // ---- output projection + residual LN1 ----
    gemm(ctx_bf, Wto, nullptr, attn_out, nullptr, MTOK, 256, 256, 0);
    ln_kernel<<<MTOK, 256, 0, stream>>>(attn_out, xt_f32, n1_g, n1_b,
                                        y1, y1_bf, 0, 0);

    // ---- FFN + residual LN2 (final, transposed store) ----
    gemm(y1_bf, Wtf1, ffn_b1, nullptr, f1_bf, MTOK, 1024, 256, 1);
    gemm(f1_bf, Wtf2, ffn_b2, f2, nullptr, MTOK, 256, 1024, 0);
    ln_kernel<<<MTOK, 256, 0, stream>>>(f2, y1, n2_g, n2_b,
                                        out, nullptr, 0, /*transpose_out=*/1);
}
